// CausalFlowModel_42477226557801
// MI455X (gfx1250) — matrix operand order repaired and checked by reading, not compiled, not run on hardware
//
#include <hip/hip_runtime.h>
#include <hip/hip_bf16.h>

typedef __attribute__((ext_vector_type(16))) _Float16 v16h;
typedef __attribute__((ext_vector_type(8)))  float    v8f;

#define B_TOT   4096
#define T_STEPS 256
#define STATE   16
#define CTRL1   9      // CTRL + 1
#define RNN     64
#define Z       80     // RNN + STATE
#define KPAD    96     // Z + CTRL1 padded to 3*32
#define G4      320    // 4*Z
#define XH      128    // x_dnn hidden
#define UH      160    // u_dnn hidden

__device__ __forceinline__ float fast_sigmoid(float x) {
    return 1.0f / (1.0f + __expf(-x));
}
__device__ __forceinline__ float fast_tanh(float x) {
    // 1 - 2/(e^{2x}+1): saturates correctly at +-1
    return 1.0f - 2.0f / (__expf(2.0f * x) + 1.0f);
}

// ---------------- encoder: z = [x, x_dnn(x)]  -> z_ws[B][80] ----------------
__launch_bounds__(128)
__global__ void enc_kernel(const float* __restrict__ x,
                           const float* __restrict__ W1, const float* __restrict__ b1,
                           const float* __restrict__ W2, const float* __restrict__ b2,
                           const float* __restrict__ W3, const float* __restrict__ b3,
                           float* __restrict__ z_ws) {
    __shared__ float xs[STATE], h1[XH], h2[XH];
    const int row = blockIdx.x, tid = threadIdx.x;
    if (tid < STATE) xs[tid] = x[row * STATE + tid];
    __syncthreads();
    {
        float a = b1[tid];
        #pragma unroll
        for (int i = 0; i < STATE; ++i) a += xs[i] * W1[i * XH + tid];
        h1[tid] = fast_tanh(a);
    }
    __syncthreads();
    {
        float a = b2[tid];
        for (int i = 0; i < XH; ++i) a += h1[i] * W2[i * XH + tid];
        h2[tid] = fast_tanh(a);
    }
    __syncthreads();
    if (tid < RNN) {
        float a = b3[tid];
        for (int i = 0; i < XH; ++i) a += h2[i] * W3[i * RNN + tid];
        z_ws[(size_t)row * Z + STATE + tid] = a;
    } else if (tid < RNN + STATE) {
        z_ws[(size_t)row * Z + (tid - RNN)] = xs[tid - RNN];
    }
}

// ---------------- LSTM scan over T with WMMA gate matmuls ----------------
// block = 128 threads (4 waves), handles 16 batch rows.
// g[16x320] = Acat[16x96] * Wcat[96x320] + bias, Wcat = [W_ih ; W_hh ; 0-pad]
// wave w owns gate columns [80w, 80w+80) == exactly gate w's 5 N-tiles.
__launch_bounds__(128, 1)
__global__ void lstm_kernel(const float* __restrict__ rnn_input,
                            const float* __restrict__ deltas,
                            const int*   __restrict__ lengths,
                            const float* __restrict__ W_ih,
                            const float* __restrict__ W_hh,
                            const float* __restrict__ b_rnn,
                            const float* __restrict__ z_ws,
                            float*       __restrict__ enc_ws) {
    __shared__ float acat[16][KPAD];   // [u(9) | h(80) | zero pad(7)] fp32
    __shared__ float g_lds[16][G4];
    __shared__ float c_lds[16][Z];
    __shared__ float hprev[16][Z];     // h at t == len-2 (init: z)
    __shared__ float hlast[16][Z];     // h at t == len-1
    __shared__ float d_step[16];
    __shared__ int   len_s[16];

    const int tid   = threadIdx.x;
    const int lane  = tid & 31;
    const int wave  = tid >> 5;        // 0..3
    const int nlane = lane & 15;
    const int hh    = lane >> 4;       // lane half (0/1)
    const int b0    = blockIdx.x * 16;

    // ---- init LDS state ----
    for (int idx = tid; idx < 16 * KPAD; idx += 128) {
        int m = idx / KPAD, k = idx % KPAD;
        float v = 0.0f;
        if (k >= CTRL1 && k < CTRL1 + Z) v = z_ws[(size_t)(b0 + m) * Z + (k - CTRL1)];
        acat[m][k] = v;
    }
    for (int idx = tid; idx < 16 * Z; idx += 128) {
        int m = idx / Z, j = idx % Z;
        float zv = z_ws[(size_t)(b0 + m) * Z + j];
        c_lds[m][j] = 0.0f;
        hprev[m][j] = zv;   // h_shift fallback when len == 1
        hlast[m][j] = zv;
    }
    if (tid < 16) len_s[tid] = lengths[b0 + tid];

    // ---- preload B (weight) fragments into VGPRs: 3 K-chunks x 5 N-tiles ----
    // B 32x16 f16 layout: lanes 0-15 hold K=0..15, lanes 16-31 hold K=16..31,
    // column n = lane&15 (contiguous k order chosen here, same as A).
    v16h Bf[3][5];
    v8f  biasf[5];
    #pragma unroll
    for (int kt = 0; kt < 3; ++kt) {
        #pragma unroll
        for (int q = 0; q < 5; ++q) {
            const int col = (wave * 5 + q) * 16 + nlane;
            v16h bfr;
            #pragma unroll
            for (int j = 0; j < 16; ++j) {
                int k = kt * 32 + hh * 16 + j;
                float w = 0.0f;
                if (k < CTRL1)              w = W_ih[k * G4 + col];
                else if (k < CTRL1 + Z)     w = W_hh[(k - CTRL1) * G4 + col];
                bfr[j] = (_Float16)w;
            }
            Bf[kt][q] = bfr;
        }
    }
    #pragma unroll
    for (int q = 0; q < 5; ++q) {
        float bv = b_rnn[(wave * 5 + q) * 16 + nlane];
        v8f c;
        #pragma unroll
        for (int v = 0; v < 8; ++v) c[v] = bv;   // bias depends on column only
        biasf[q] = c;
    }
    __syncthreads();

    // ---- serial scan ----
    for (int t = 0; t < T_STEPS; ++t) {
        // stage u_t (16 rows x 9) into acat[:,0:9], delta_t into d_step
        for (int idx = tid; idx < 16 * CTRL1; idx += 128) {
            int m = idx / CTRL1, k = idx % CTRL1;
            acat[m][k] = rnn_input[((size_t)(b0 + m) * T_STEPS + t) * CTRL1 + k];
            if (t + 1 < T_STEPS)
                __builtin_prefetch(&rnn_input[((size_t)(b0 + m) * T_STEPS + t + 1) * CTRL1 + k], 0, 3);
        }
        if (tid < 16) d_step[tid] = deltas[(size_t)(b0 + tid) * T_STEPS + t];
        __syncthreads();

        // build 16x32 A fragments in contiguous k order (same as B):
        // lane holds row m=lane&15; element j -> K = half*16 + j
        v16h Af[3];
        const int m = nlane;
        #pragma unroll
        for (int kt = 0; kt < 3; ++kt) {
            v16h a;
            #pragma unroll
            for (int j = 0; j < 16; ++j) {
                int v = j >> 1, e = j & 1;
                int k = kt * 32 + (hh << 4) + j;
                a[j] = (_Float16)acat[m][k];
            }
            Af[kt] = a;
        }

        // 15 chained WMMAs: g_tile = A*B + bias (fp32 accumulate)
        v8f acc[5];
        #pragma unroll
        for (int q = 0; q < 5; ++q) {
            v8f c = biasf[q];
            c = __builtin_amdgcn_wmma_f32_16x16x32_f16(false, Af[0], false, Bf[0][q], (short)0, c, false, false);
            c = __builtin_amdgcn_wmma_f32_16x16x32_f16(false, Af[1], false, Bf[1][q], (short)0, c, false, false);
            c = __builtin_amdgcn_wmma_f32_16x16x32_f16(false, Af[2], false, Bf[2][q], (short)0, c, false, false);
            acc[q] = c;
        }
        // scatter D per f32 C/D layout: lane<16 -> rows v, lane>=16 -> rows v+8
        #pragma unroll
        for (int q = 0; q < 5; ++q) {
            const int col = (wave * 5 + q) * 16 + nlane;
            #pragma unroll
            for (int v = 0; v < 8; ++v)
                g_lds[v + hh * 8][col] = acc[q][v];
        }
        __syncthreads();

        // elementwise gate math: 16*80 = 1280 elements, 10 per thread
        #pragma unroll
        for (int r = 0; r < 10; ++r) {
            int idx = r * 128 + tid;
            int mm = idx / Z, j = idx % Z;
            float fi = fast_sigmoid(g_lds[mm][j]);
            float ff = fast_sigmoid(g_lds[mm][Z + j]);
            float gg = fast_tanh   (g_lds[mm][2 * Z + j]);
            float fo = fast_sigmoid(g_lds[mm][3 * Z + j]);
            float cn = ff * c_lds[mm][j] + fi * gg;
            float ht = fo * fast_tanh(cn);
            float ho = acat[mm][CTRL1 + j];
            float hn = ho + d_step[mm] * (ht - ho);
            c_lds[mm][j] = cn;
            acat[mm][CTRL1 + j] = hn;
            int L = len_s[mm];
            if (t == L - 2) hprev[mm][j] = hn;
            if (t == L - 1) hlast[mm][j] = hn;
        }
        __syncthreads();
    }

    // encoded at last valid step: (1-d)*h[len-2 or z] + d*h[len-1]
    #pragma unroll
    for (int r = 0; r < 10; ++r) {
        int idx = r * 128 + tid;
        int mm = idx / Z, j = idx % Z;
        int L = len_s[mm];
        float dl = deltas[(size_t)(b0 + mm) * T_STEPS + (L - 1)];
        enc_ws[(size_t)(b0 + mm) * Z + j] = (1.0f - dl) * hprev[mm][j] + dl * hlast[mm][j];
    }
}

// ---------------- decoder: out = u_dnn(encoded_last)[:, :16] ----------------
__launch_bounds__(160)
__global__ void dec_kernel(const float* __restrict__ enc_ws,
                           const float* __restrict__ W1, const float* __restrict__ b1,
                           const float* __restrict__ W2, const float* __restrict__ b2,
                           const float* __restrict__ W3, const float* __restrict__ b3,
                           float* __restrict__ out) {
    __shared__ float e0[Z], h1[UH], h2[UH];
    const int row = blockIdx.x, tid = threadIdx.x;
    if (tid < Z) e0[tid] = enc_ws[(size_t)row * Z + tid];
    __syncthreads();
    {
        float a = b1[tid];
        for (int i = 0; i < Z; ++i) a += e0[i] * W1[i * UH + tid];
        h1[tid] = fast_tanh(a);
    }
    __syncthreads();
    {
        float a = b2[tid];
        for (int i = 0; i < UH; ++i) a += h1[i] * W2[i * UH + tid];
        h2[tid] = fast_tanh(a);
    }
    __syncthreads();
    if (tid < STATE) {
        float a = b3[tid];
        for (int i = 0; i < UH; ++i) a += h2[i] * W3[i * STATE + tid];
        out[(size_t)row * STATE + tid] = a;
    }
    // coefficients = zeros(2,2), appended after out in d_out
    if (row == 0 && tid < 4) out[(size_t)B_TOT * STATE + tid] = 0.0f;
}

extern "C" void kernel_launch(void* const* d_in, const int* in_sizes, int n_in,
                              void* d_out, int out_size, void* d_ws, size_t ws_size,
                              hipStream_t stream) {
    const float* x         = (const float*)d_in[0];
    const float* rnn_input = (const float*)d_in[1];
    const float* deltas    = (const float*)d_in[2];
    const int*   lengths   = (const int*)  d_in[3];
    const float* xW1 = (const float*)d_in[4],  *xb1 = (const float*)d_in[5];
    const float* xW2 = (const float*)d_in[6],  *xb2 = (const float*)d_in[7];
    const float* xW3 = (const float*)d_in[8],  *xb3 = (const float*)d_in[9];
    const float* uW1 = (const float*)d_in[10], *ub1 = (const float*)d_in[11];
    const float* uW2 = (const float*)d_in[12], *ub2 = (const float*)d_in[13];
    const float* uW3 = (const float*)d_in[14], *ub3 = (const float*)d_in[15];
    const float* W_ih  = (const float*)d_in[16];
    const float* W_hh  = (const float*)d_in[17];
    const float* b_rnn = (const float*)d_in[18];

    float* z_ws   = (float*)d_ws;                      // [B][80]
    float* enc_ws = z_ws + (size_t)B_TOT * Z;          // [B][80]
    float* out    = (float*)d_out;                     // [B][16] + 4 coeff

    enc_kernel<<<B_TOT, XH, 0, stream>>>(x, xW1, xb1, xW2, xb2, xW3, xb3, z_ws);
    lstm_kernel<<<B_TOT / 16, 128, 0, stream>>>(rnn_input, deltas, lengths,
                                                W_ih, W_hh, b_rnn, z_ws, enc_ws);
    dec_kernel<<<B_TOT, UH, 0, stream>>>(enc_ws, uW1, ub1, uW2, ub2, uW3, ub3, out);
}
